// AdaptiveLRLinearWithChannel_36481452212318
// MI455X (gfx1250) — matrix operand order repaired and checked by reading, not compiled, not run on hardware
//
#include <hip/hip_runtime.h>
#include <hip/hip_bf16.h>

// MI455X / gfx1250, wave32. BF16 WMMA GEMM with f32 accumulation, LDS-staged B,
// sched_barrier-pinned software pipeline, async global->LDS staging, NT stores.
//
// out[b] = x[b] @ W_b + bias[idx_b],  W_b = (U[idx_b] @ V).reshape(256,256)
//   x: [256, 2048, 256] f32   indices: [256] i32
//   U: [1024, 4] f32          V: [4, 65536] f32      bias: [1024, 1, 256] f32
//   out: [256, 2048, 256] f32
//
// Roofline: 68.7 GFLOP vs ~1 GB HBM -> ~45us memory floor @ 23.3 TB/s; bf16
// WMMA (16x16x32, f32 accum) puts compute under the floor -> memory-bound.
//
// d_ws layout (32 MB, L2-resident): per batch, fragment (ko,nt) occupies 1 KB:
//   half-chunk 0: lanes' bf16[8] for elements e=0..7  at  byte lane*16
//   half-chunk 1: lanes' bf16[8] for elements e=8..15 at  byte 512+lane*16
//   element (lane,e) = W[k = ko*32 + (lane<16?0:16) + e][n = nt*16 + (lane&15)]

typedef __bf16 bf16_t;
typedef __attribute__((ext_vector_type(16))) __bf16 v16bf;
typedef __attribute__((ext_vector_type(8)))  __bf16 v8bf;
typedef __attribute__((ext_vector_type(8)))  float  v8f;

#define B_SZ    256
#define N_SEQ   2048
#define IN_SZ   256
#define OUT_SZ  256
#define RANK    4

#if defined(__has_builtin)
#  if __has_builtin(__builtin_amdgcn_global_load_async_to_lds_b128) && \
      __has_builtin(__builtin_amdgcn_s_wait_asynccnt)
#    define USE_ASYNC_LDS 1
#  endif
#  if __has_builtin(__builtin_amdgcn_sched_barrier)
#    define SCHED_FENCE() __builtin_amdgcn_sched_barrier(0)
#  endif
#endif
#ifndef SCHED_FENCE
#  define SCHED_FENCE()
#endif

#if defined(USE_ASYNC_LDS)
typedef __attribute__((vector_size(4 * sizeof(int)))) int b128_t;
typedef __attribute__((address_space(1))) b128_t* gp_b128;
typedef __attribute__((address_space(3))) b128_t* lp_b128;
#endif

// ---------------------------------------------------------------------------
// Kernel 1: weight synthesis into WMMA-B swizzled bf16 layout (above).
// grid = (16, 256), block = 256.  thread t = ko*512 + nt*32 + lane.
// ---------------------------------------------------------------------------
__global__ __launch_bounds__(256) void lrw_synth_kernel(
    const int* __restrict__ indices,
    const float* __restrict__ U,
    const float* __restrict__ V,
    bf16_t* __restrict__ wsW)
{
    const int b    = blockIdx.y;
    const int t    = blockIdx.x * blockDim.x + threadIdx.x;   // 0..4095
    const int lane = t & 31;
    const int nt   = (t >> 5) & 15;
    const int ko   = t >> 9;                                  // 0..7

    const int idx = indices[b];
    const float u0 = U[idx * RANK + 0];
    const float u1 = U[idx * RANK + 1];
    const float u2 = U[idx * RANK + 2];
    const float u3 = U[idx * RANK + 3];

    const int k0 = ko * 32 + ((lane & 16) ? 16 : 0);
    const int n  = nt * 16 + (lane & 15);

    v8bf lo, hiv;
#pragma unroll
    for (int e = 0; e < 16; ++e) {
        const int k = k0 + e;
        const float* vp = V + (size_t)k * OUT_SZ + n;        // V[r][k*256+n]
        const float wv = u0 * vp[0]
                       + u1 * vp[65536]
                       + u2 * vp[131072]
                       + u3 * vp[196608];
        if (e < 8) lo[e] = (bf16_t)wv; else hiv[e - 8] = (bf16_t)wv;
    }

    bf16_t* dst = wsW + ((size_t)b << 16)
                      + (size_t)(ko * 16 + nt) * 512 + lane * 8;
    *(v8bf*)dst         = lo;
    *(v8bf*)(dst + 256) = hiv;
}

// ---------------------------------------------------------------------------
// helpers
// ---------------------------------------------------------------------------
__device__ __forceinline__ v16bf cvt_a(float4 f0, float4 f1, float4 f2, float4 f3)
{
    v16bf a;
    a[0]  = (bf16_t)f0.x;  a[1]  = (bf16_t)f0.y;
    a[2]  = (bf16_t)f0.z;  a[3]  = (bf16_t)f0.w;
    a[4]  = (bf16_t)f1.x;  a[5]  = (bf16_t)f1.y;
    a[6]  = (bf16_t)f1.z;  a[7]  = (bf16_t)f1.w;
    a[8]  = (bf16_t)f2.x;  a[9]  = (bf16_t)f2.y;
    a[10] = (bf16_t)f2.z;  a[11] = (bf16_t)f2.w;
    a[12] = (bf16_t)f3.x;  a[13] = (bf16_t)f3.y;
    a[14] = (bf16_t)f3.z;  a[15] = (bf16_t)f3.w;
    return a;
}

__device__ __forceinline__ v16bf load_b(const bf16_t* lb, int frag)
{
    union { v16bf v; v8bf h[2]; } bu;
    const bf16_t* p = lb + frag * 512;
    bu.h[0] = *(const v8bf*)p;           // elements 0..7
    bu.h[1] = *(const v8bf*)(p + 256);   // elements 8..15
    return bu.v;
}

// ---------------------------------------------------------------------------
// Kernel 2: per-batch GEMM. grid = (32, 256): gx = mchunk*2 + nhalf.
// Block: 8 waves = 128 rows x 128 cols; W half-tile (64 KB) staged in LDS.
// ---------------------------------------------------------------------------
__global__ __launch_bounds__(256) void lrw_gemm_kernel(
    const float* __restrict__ x,
    const int* __restrict__ indices,
    const bf16_t* __restrict__ wsW,
    const float* __restrict__ bias,
    float* __restrict__ out)
{
    __shared__ bf16_t wtile[8 * 8 * 512];   // 64 KB: frag(ko,ntl) at (ko*8+ntl)*512

    const int b    = blockIdx.y;
    const int gx   = blockIdx.x;
    const int nh   = gx & 1;                 // N half (0/1)
    const int mc   = gx >> 1;                // M chunk (0..15)
    const int tid  = threadIdx.x;
    const int lane = tid & 31;               // wave32
    const int wave = tid >> 5;
    const int row0 = (mc * 8 + wave) * 16;   // strip base (M)
    const int rlo  = lane & 15;
    const bool hi  = lane >= 16;

    const bf16_t* __restrict__ wB = wsW + ((size_t)b << 16);

    // ---- cooperative stage: 64 KB = 4096 x 16B chunks, 16 per thread ------
#pragma unroll
    for (int j = 0; j < 16; ++j) {
        const int c  = j * 256 + tid;
        const int f  = c >> 6;               // local frag  (ko*8+ntl)
        const int w  = c & 63;               // 16B slot within frag
        const int ko = f >> 3;
        const int nl = f & 7;
        const int g  = ko * 16 + nh * 8 + nl;
        const bf16_t* src = wB + (size_t)g * 512 + w * 8;
        bf16_t*       dst = &wtile[(size_t)c * 8];
#if defined(USE_ASYNC_LDS)
        __builtin_amdgcn_global_load_async_to_lds_b128(
            (gp_b128)src, (lp_b128)dst, 0, 0);
#else
        *(float4*)dst = *(const float4*)src;
#endif
    }
#if defined(USE_ASYNC_LDS)
    __builtin_amdgcn_s_wait_asynccnt(0);
#endif
    __syncthreads();

    // ---- sched_barrier-pinned MMA pipeline --------------------------------
    // A layout: lanes 0-15: K = ko*32 + {0..15}; lanes 16-31: +16.
    const float* __restrict__ xrow =
        x + ((size_t)b * N_SEQ + (row0 + rlo)) * IN_SZ;
    const int klo = hi ? 16 : 0;

    v8f acc[8] = {};
    const bf16_t* lb = &wtile[lane * 8];

    // prologue: A frag 0 only (one-time serialized cvt is fine)
    float4 r0 = *(const float4*)(xrow + klo);
    float4 r1 = *(const float4*)(xrow + klo + 4);
    float4 r2 = *(const float4*)(xrow + klo + 8);
    float4 r3 = *(const float4*)(xrow + klo + 12);
    v16bf aCur = cvt_a(r0, r1, r2, r3);

#pragma unroll
    for (int ko = 0; ko < 8; ++ko) {
        // (1) issue next-A global loads: consumed only after 8 WMMAs
        float4 n0, n1, n2, n3;
        if (ko < 7) {
            const int kb = (ko + 1) * 32 + klo;
            n0 = *(const float4*)(xrow + kb);
            n1 = *(const float4*)(xrow + kb + 4);
            n2 = *(const float4*)(xrow + kb + 8);
            n3 = *(const float4*)(xrow + kb + 12);
        }
        // (2) issue ALL 16 ds_loads for this ko's 8 B fragments
        v16bf bf[8];
#pragma unroll
        for (int nl = 0; nl < 8; ++nl)
            bf[nl] = load_b(lb, ko * 8 + nl);

        // (3) fence: loads may not sink below; WMMAs may not rise above
        SCHED_FENCE();

        // (4) 8 WMMAs drain the fragment queue with partial dscnt waits
#pragma unroll
        for (int nl = 0; nl < 8; ++nl)
            acc[nl] = __builtin_amdgcn_wmma_f32_16x16x32_bf16(
                false, aCur, false, bf[nl], (short)0, acc[nl], false, false);

        // (5) convert next A (single wait, full-chain slack behind it)
        if (ko < 7) aCur = cvt_a(n0, n1, n2, n3);
    }

    // ---- epilogue: D layout = VGPR v -> row (hi?8:0)+v, col = lane&15 ------
    const int idx   = indices[b];
    const int rbase = row0 + (hi ? 8 : 0);
#pragma unroll
    for (int nl = 0; nl < 8; ++nl) {
        const int col = (nh * 8 + nl) * 16 + rlo;
        const float bv = bias[(size_t)idx * OUT_SZ + col];
        float* __restrict__ op =
            out + ((size_t)b * N_SEQ + rbase) * OUT_SZ + col;
#pragma unroll
        for (int v = 0; v < 8; ++v)
            __builtin_nontemporal_store(acc[nl][v] + bv, op + (size_t)v * OUT_SZ);
    }
}

// ---------------------------------------------------------------------------
extern "C" void kernel_launch(void* const* d_in, const int* in_sizes, int n_in,
                              void* d_out, int out_size, void* d_ws, size_t ws_size,
                              hipStream_t stream) {
    const float* x       = (const float*)d_in[0];
    const int*   indices = (const int*)  d_in[1];
    const float* U       = (const float*)d_in[2];
    const float* V       = (const float*)d_in[3];
    const float* bias    = (const float*)d_in[4];
    float*       out     = (float*)d_out;
    bf16_t*      wsW     = (bf16_t*)d_ws;   // needs 32 MB

    lrw_synth_kernel<<<dim3(16, B_SZ), 256, 0, stream>>>(indices, U, V, wsW);
    lrw_gemm_kernel<<<dim3(32, B_SZ), 256, 0, stream>>>(x, indices, wsW, bias, out);
}
